// TextMaskedMultiheadSelfAttention_83416854823051
// MI455X (gfx1250) — matrix operand order repaired and checked by reading, not compiled, not run on hardware
//
#include <hip/hip_runtime.h>
#include <stdint.h>

#define D_MODEL  1024
#define N_HEADS  16
#define HEAD_DIM 64
#define SEQ_T    2048
#define BATCH    2
#define ROWS     (BATCH * SEQ_T)   // 4096

typedef _Float16 v16h __attribute__((ext_vector_type(16)));
typedef _Float16 v8h  __attribute__((ext_vector_type(8)));
typedef float    v8f  __attribute__((ext_vector_type(8)));
typedef unsigned int u32x4 __attribute__((ext_vector_type(4)));
typedef int          i32x8 __attribute__((ext_vector_type(8)));
typedef int          i32x4 __attribute__((ext_vector_type(4)));

#if __has_builtin(__builtin_amdgcn_tensor_load_to_lds) && \
    __has_builtin(__builtin_amdgcn_s_wait_tensorcnt)
#define USE_TDM 1
#else
#define USE_TDM 0
#endif

// -------------------------------------------------------------------------
// Kernel 1: LayerNorm (f32 in -> f16 normalized out), one row per block
// -------------------------------------------------------------------------
__global__ void __launch_bounds__(256)
ln_f16_kernel(const float* __restrict__ x,
              const float* __restrict__ g,
              const float* __restrict__ bta,
              _Float16* __restrict__ xn) {
    __shared__ float ssum[256], ssq[256];
    int row = blockIdx.x;
    const float* xr = x + (size_t)row * D_MODEL;
    float vals[4];
    float s = 0.f, sq = 0.f;
#pragma unroll
    for (int i = 0; i < 4; ++i) {
        float v = xr[threadIdx.x + i * 256];
        vals[i] = v; s += v; sq += v * v;
    }
    ssum[threadIdx.x] = s; ssq[threadIdx.x] = sq;
    __syncthreads();
    for (int off = 128; off > 0; off >>= 1) {
        if (threadIdx.x < off) {
            ssum[threadIdx.x] += ssum[threadIdx.x + off];
            ssq[threadIdx.x]  += ssq[threadIdx.x + off];
        }
        __syncthreads();
    }
    float mu  = ssum[0] * (1.0f / D_MODEL);
    float var = ssq[0] * (1.0f / D_MODEL) - mu * mu;
    float rs  = rsqrtf(var + 1e-5f);
    _Float16* xnr = xn + (size_t)row * D_MODEL;
#pragma unroll
    for (int i = 0; i < 4; ++i) {
        int c = threadIdx.x + i * 256;
        xnr[c] = (_Float16)((vals[i] - mu) * rs * g[c] + bta[c]);
    }
}

// -------------------------------------------------------------------------
// Kernel 2: f32 -> f16 conversion (weights)
// -------------------------------------------------------------------------
__global__ void f32_to_f16_kernel(const float* __restrict__ src,
                                  _Float16* __restrict__ dst, int n) {
    int i = blockIdx.x * blockDim.x + threadIdx.x;
    if (i < n) dst[i] = (_Float16)src[i];
}

// -------------------------------------------------------------------------
// Kernel 3: y = xn @ W^T + b  via v_wmma_f32_16x16x32_f16.
// Each wave: 16 rows x 64 cols output tile.
// transposed==0: store (B,H,T,hd);  transposed==1: store (B,H,hd,T)
// -------------------------------------------------------------------------
__global__ void __launch_bounds__(256)
qkv_wmma_kernel(const _Float16* __restrict__ xn,
                const _Float16* __restrict__ w,      // (D_out, D_in) row-major
                const float* __restrict__ bias,
                _Float16* __restrict__ dst,
                int transposed) {
    int lane = threadIdx.x & 31;
    int wid  = blockIdx.x * (blockDim.x >> 5) + (threadIdx.x >> 5);
    int tm   = wid >> 4;         // 256 row-tiles
    int tn4  = wid & 15;         // 16 col-macrotiles of 64
    int m0   = tm * 16;
    int n0   = tn4 * 64;

    int lr  = lane & 15;
    int lko = (lane & 16) ? 8 : 0;     // unused; A-fragment also uses bko
    int bko = (lane & 16) ? 16 : 0;    // B-fragment k offset

    v8f acc[4] = {};

    for (int kk = 0; kk < D_MODEL; kk += 32) {
        v16h a;
        {
            const _Float16* ap = xn + (size_t)(m0 + lr) * D_MODEL + kk + bko;
            v8h lo = *(const v8h*)(ap);
            v8h hi = *(const v8h*)(ap + 8);
#pragma unroll
            for (int i = 0; i < 8; ++i) { a[i] = lo[i]; a[i + 8] = hi[i]; }
        }
#pragma unroll
        for (int c = 0; c < 4; ++c) {
            const _Float16* bp =
                w + (size_t)(n0 + c * 16 + lr) * D_MODEL + kk + bko;
            v16h b = *(const v16h*)bp;
            acc[c] = __builtin_amdgcn_wmma_f32_16x16x32_f16(
                false, a, false, b, (short)0, acc[c], false, false);
        }
    }

#pragma unroll
    for (int c = 0; c < 4; ++c) {
        int n = n0 + c * 16 + lr;
        float bv = bias[n];
        int h = n >> 6, d = n & (HEAD_DIM - 1);
#pragma unroll
        for (int r = 0; r < 8; ++r) {
            int m = m0 + r + ((lane & 16) ? 8 : 0);
            int bi = m >> 11, t = m & (SEQ_T - 1);
            _Float16 val = (_Float16)(acc[c][r] + bv);
            if (transposed)
                dst[(((size_t)bi * N_HEADS + h) * HEAD_DIM + d) * SEQ_T + t] = val;
            else
                dst[(((size_t)bi * N_HEADS + h) * SEQ_T + t) * HEAD_DIM + d] = val;
        }
    }
}

// -------------------------------------------------------------------------
// TDM helper: DMA a 32x64 f16 tile (row stride 64 elements) into LDS.
// Descriptor bit layout per CDNA5 ISA §8.3-8.5. 6-arg builtin form
// (clang-23 / therock signature): (v4u32, v8i32, v4i32, v4i32, v8i32, i32).
// -------------------------------------------------------------------------
#if USE_TDM
__device__ __forceinline__ void tdm_load_tile32x64(const _Float16* gsrc,
                                                   uint32_t lds_off) {
    uint64_t ga = (uint64_t)(uintptr_t)gsrc;
    u32x4 g0;
    g0[0] = 1u;                                            // count = 1 valid D#
    g0[1] = lds_off;                                       // lds_addr
    g0[2] = (uint32_t)ga;                                  // global_addr[31:0]
    g0[3] = ((uint32_t)(ga >> 32) & 0x01FFFFFFu)           // global_addr[56:32]
            | 0x80000000u;                                 // type = 2 ("image")
    i32x8 g1;
    g1[0] = 0x00010000;          // workgroup_mask=0, data_size=1 (2 bytes)
    g1[1] = (64 << 16);          // tensor_dim0 = 64
    g1[2] = (2048 << 16);        // tensor_dim1 = 2048
    g1[3] = (64 << 16);          // tile_dim0 = 64
    g1[4] = 32;                  // tile_dim1 = 32, tile_dim2 = 0
    g1[5] = 64;                  // tensor_dim0_stride = 64
    g1[6] = 0;                   // tensor_dim1_stride = 0 (2D tile)
    g1[7] = 0;
    i32x4 gz4 = {0, 0, 0, 0};
    i32x8 gz8 = {0, 0, 0, 0, 0, 0, 0, 0};
    __builtin_amdgcn_tensor_load_to_lds(g0, g1, gz4, gz4, gz8, 0);
}
#endif

// -------------------------------------------------------------------------
// Kernel 4: flash attention. One wave = 16 query rows of one (b,h).
// K tiles staged into LDS by the Tensor Data Mover, double-buffered.
// V is pre-transposed (B,H,hd,T) so its B-fragments are contiguous loads.
// out = x + softmax(Q K^T / sqrt(T)) V   (no mask, per reference)
// -------------------------------------------------------------------------
#define ATTN_WAVES 4

__global__ void __launch_bounds__(128)
attn_kernel(const _Float16* __restrict__ qb,   // (B,H,T,hd)
            const _Float16* __restrict__ kb,   // (B,H,T,hd)
            const _Float16* __restrict__ vt,   // (B,H,hd,T)  transposed!
            const float* __restrict__ x,
            float* __restrict__ out) {
    __shared__ _Float16 ktile[ATTN_WAVES][2][32 * HEAD_DIM]; // 32 KB
    __shared__ _Float16 plds[ATTN_WAVES][16][32];            //  4 KB
    int lane = threadIdx.x & 31;
    int wv   = threadIdx.x >> 5;
    int wid  = blockIdx.x * ATTN_WAVES + wv;
    int tq   = wid & 127;                  // T/16 = 128 query tiles
    int bh   = wid >> 7;                   // 0..31  (b*H + h)
    int t0   = tq * 16;

    const size_t base = (size_t)bh * SEQ_T * HEAD_DIM;
    const _Float16* Q = qb + base;
    const _Float16* K = kb + base;
    const _Float16* V = vt + base;         // (hd, T) per head

    int lr  = lane & 15;
    int lko = (lane & 16) ? 8 : 0;
    int bko = (lane & 16) ? 16 : 0;

    // Q fragments: 16 rows x 64 dims = 2 A-fragments (K=32 each)
    v16h qf[2];
#pragma unroll
    for (int c = 0; c < 2; ++c) {
        const _Float16* ap = Q + (size_t)(t0 + lr) * HEAD_DIM + c * 32 + bko;
        v8h lo = *(const v8h*)ap;
        v8h hi = *(const v8h*)(ap + 8);
#pragma unroll
        for (int i = 0; i < 8; ++i) { qf[c][i] = lo[i]; qf[c][i + 8] = hi[i]; }
    }

    v8f o[4] = {};
    float m[8], l[8];
#pragma unroll
    for (int r = 0; r < 8; ++r) { m[r] = -1e30f; l[r] = 0.f; }

    const float scale = 0.022097086912079612f;   // 1/sqrt(2048)
    const int NTILES = SEQ_T / 32;               // 64

#if USE_TDM
    // prologue: DMA first K tile into buffer 0
    tdm_load_tile32x64(K, (uint32_t)(uintptr_t)&ktile[wv][0][0]);
#endif

    for (int jt = 0; jt < NTILES; ++jt) {
        int j0  = jt * 32;
        int cur = jt & 1;
#if USE_TDM
        // prefetch next K tile into the other buffer via TDM
        if (jt + 1 < NTILES)
            tdm_load_tile32x64(K + (size_t)(j0 + 32) * HEAD_DIM,
                               (uint32_t)(uintptr_t)&ktile[wv][cur ^ 1][0]);
        // wait for current tile (allow the prefetch to stay in flight)
        if (jt + 1 < NTILES) __builtin_amdgcn_s_wait_tensorcnt(1);
        else                 __builtin_amdgcn_s_wait_tensorcnt(0);
#endif
        // ---- scores: S = Q K^T for 32 keys (two 16x16 tiles) ----
        v8f s0 = {}, s1 = {};
#pragma unroll
        for (int c = 0; c < 2; ++c) {
#if USE_TDM
            v16h b0 = *(const v16h*)&ktile[wv][cur][(size_t)lr * HEAD_DIM + c * 32 + bko];
            v16h b1 = *(const v16h*)&ktile[wv][cur][(size_t)(16 + lr) * HEAD_DIM + c * 32 + bko];
#else
            v16h b0 = *(const v16h*)(K + (size_t)(j0 + lr) * HEAD_DIM + c * 32 + bko);
            v16h b1 = *(const v16h*)(K + (size_t)(j0 + 16 + lr) * HEAD_DIM + c * 32 + bko);
#endif
            s0 = __builtin_amdgcn_wmma_f32_16x16x32_f16(
                false, qf[c], false, b0, (short)0, s0, false, false);
            s1 = __builtin_amdgcn_wmma_f32_16x16x32_f16(
                false, qf[c], false, b1, (short)0, s1, false, false);
        }
        // ---- online softmax over the 32 keys ----
        float p0[8], p1[8];
#pragma unroll
        for (int r = 0; r < 8; ++r) {
            float a0 = s0[r] * scale;
            float a1 = s1[r] * scale;
            float mt = fmaxf(a0, a1);
#pragma unroll
            for (int msk = 1; msk < 16; msk <<= 1)
                mt = fmaxf(mt, __shfl_xor(mt, msk, 32));
            float mn    = fmaxf(m[r], mt);
            float alpha = __expf(m[r] - mn);
            float e0 = __expf(a0 - mn);
            float e1 = __expf(a1 - mn);
            float rs = e0 + e1;
#pragma unroll
            for (int msk = 1; msk < 16; msk <<= 1)
                rs += __shfl_xor(rs, msk, 32);
            l[r] = l[r] * alpha + rs;
            m[r] = mn;
            p0[r] = e0; p1[r] = e1;
#pragma unroll
            for (int c = 0; c < 4; ++c) o[c][r] *= alpha;
        }
        // ---- repack P (C-layout -> A-layout) through LDS ----
#pragma unroll
        for (int r = 0; r < 8; ++r) {
            int row = r + ((lane & 16) ? 8 : 0);
            plds[wv][row][lr]      = (_Float16)p0[r];
            plds[wv][row][16 + lr] = (_Float16)p1[r];
        }
        __syncthreads();
        v16h pa;
        {
            const _Float16* ap = &plds[wv][lr][bko];
            v8h lo = *(const v8h*)ap;
            v8h hi = *(const v8h*)(ap + 8);
#pragma unroll
            for (int i = 0; i < 8; ++i) { pa[i] = lo[i]; pa[i + 8] = hi[i]; }
        }
        __syncthreads();
        // ---- O += P @ V: V^T layout makes B-fragments contiguous ----
#pragma unroll
        for (int c = 0; c < 4; ++c) {
            v16h bv = *(const v16h*)(V + (size_t)(c * 16 + lr) * SEQ_T + j0 + bko);
            o[c] = __builtin_amdgcn_wmma_f32_16x16x32_f16(
                false, pa, false, bv, (short)0, o[c], false, false);
        }
    }

    // ---- epilogue: normalize + residual, write f32 output ----
    int b = bh >> 4, h = bh & 15;
#pragma unroll
    for (int c = 0; c < 4; ++c) {
        int d = h * HEAD_DIM + c * 16 + lr;
#pragma unroll
        for (int r = 0; r < 8; ++r) {
            int t = t0 + r + ((lane & 16) ? 8 : 0);
            size_t idx = ((size_t)b * SEQ_T + t) * D_MODEL + d;
            out[idx] = x[idx] + o[c][r] / l[r];
        }
    }
}

// -------------------------------------------------------------------------
// Host-side launch
// -------------------------------------------------------------------------
extern "C" void kernel_launch(void* const* d_in, const int* in_sizes, int n_in,
                              void* d_out, int out_size, void* d_ws, size_t ws_size,
                              hipStream_t stream) {
    (void)in_sizes; (void)n_in; (void)out_size; (void)ws_size;
    const float* x    = (const float*)d_in[0];
    const float* wq_w = (const float*)d_in[1];
    const float* wq_b = (const float*)d_in[2];
    const float* wk_w = (const float*)d_in[3];
    const float* wk_b = (const float*)d_in[4];
    const float* wv_w = (const float*)d_in[5];
    const float* wv_b = (const float*)d_in[6];
    const float* ln_g = (const float*)d_in[7];
    const float* ln_b = (const float*)d_in[8];
    float* out = (float*)d_out;

    // workspace layout (f16 buffers), total 38 MiB
    const size_t MiB = 1ull << 20;
    char* ws = (char*)d_ws;
    _Float16* xn  = (_Float16*)(ws);                 //  8 MiB: 4096x1024
    _Float16* wqh = (_Float16*)(ws +  8 * MiB);      //  2 MiB
    _Float16* wkh = (_Float16*)(ws + 10 * MiB);      //  2 MiB
    _Float16* wvh = (_Float16*)(ws + 12 * MiB);      //  2 MiB
    _Float16* qh  = (_Float16*)(ws + 14 * MiB);      //  8 MiB (B,H,T,hd)
    _Float16* kh  = (_Float16*)(ws + 22 * MiB);      //  8 MiB (B,H,T,hd)
    _Float16* vh  = (_Float16*)(ws + 30 * MiB);      //  8 MiB (B,H,hd,T)

    // 1) LayerNorm -> f16
    ln_f16_kernel<<<ROWS, 256, 0, stream>>>(x, ln_g, ln_b, xn);

    // 2) convert weights to f16
    const int WN = D_MODEL * D_MODEL;
    f32_to_f16_kernel<<<(WN + 255) / 256, 256, 0, stream>>>(wq_w, wqh, WN);
    f32_to_f16_kernel<<<(WN + 255) / 256, 256, 0, stream>>>(wk_w, wkh, WN);
    f32_to_f16_kernel<<<(WN + 255) / 256, 256, 0, stream>>>(wv_w, wvh, WN);

    // 3) QKV projections (V stored transposed per-head)
    qkv_wmma_kernel<<<512, 256, 0, stream>>>(xn, wqh, wq_b, qh, 0);
    qkv_wmma_kernel<<<512, 256, 0, stream>>>(xn, wkh, wk_b, kh, 0);
    qkv_wmma_kernel<<<512, 256, 0, stream>>>(xn, wvh, wv_b, vh, 1);

    // 4) flash attention + residual: 4096 waves, 4 waves/block
    attn_kernel<<<1024, 128, 0, stream>>>(qh, kh, vh, x, out);
}
